// CrossAttention_82033875354221
// MI455X (gfx1250) — matrix operand order repaired and checked by reading, not compiled, not run on hardware
//
#include <hip/hip_runtime.h>

// CrossAttention (channel attention) for MI455X / gfx1250.
// All contractions on v_wmma_f32_16x16x32_bf16 (f32 accumulate).
// Workspace: 5x bf16 weights (128KB) + 4x [8][256][9216] bf16 (36MB each)
//            + scores f32 (2MB) + attn bf16 (1MB) ~= 155 MB.

#define CH  256
#define NSP 9216
#define NB  8

typedef __attribute__((ext_vector_type(4)))  unsigned int u32x4;
typedef __attribute__((ext_vector_type(8)))  unsigned int u32x8;
typedef __attribute__((ext_vector_type(4)))  int          i32x4;
typedef __attribute__((ext_vector_type(8)))  int          i32x8;
typedef __attribute__((ext_vector_type(16))) __bf16       bf16x16;
typedef __attribute__((ext_vector_type(8)))  float        f32x8;

__device__ __forceinline__ bf16x16 make_frag(u32x4 lo, u32x4 hi) {
  u32x8 w;
  w[0] = lo[0]; w[1] = lo[1]; w[2] = lo[2]; w[3] = lo[3];
  w[4] = hi[0]; w[5] = hi[1]; w[6] = hi[2]; w[7] = hi[3];
  return __builtin_bit_cast(bf16x16, w);
}

// A fragment 16x32 (MxK), source row-major [M][K].
// lanes 0-15: M=lane, K = kb+0..7 / kb+16..23 ; lanes 16-31: K = kb+8..15 / kb+24..31.
__device__ __forceinline__ bf16x16 load_frag_a(const __bf16* base, int ld,
                                               int m0, int kb, int lane) {
  int hf = lane >> 4, r = lane & 15;
  const __bf16* row = base + (long long)(m0 + r) * ld + kb + hf * 8;
  return make_frag(*(const u32x4*)row, *(const u32x4*)(row + 16));
}

// B fragment 32x16 (KxN), source stored [N][K] row-major.
// lanes 0-15: N=lane, K = kb+0..7 / kb+16..23 ; lanes 16-31: K = kb+8..15 / kb+24..31.
__device__ __forceinline__ bf16x16 load_frag_b(const __bf16* base, int ld,
                                               int n0, int kb, int lane) {
  int hf = lane >> 4, r = lane & 15;
  const __bf16* row = base + (long long)(n0 + r) * ld + kb + hf * 8;
  return make_frag(*(const u32x4*)row, *(const u32x4*)(row + 16));
}

__device__ __forceinline__ f32x8 wmma_bf16(bf16x16 a, bf16x16 b, f32x8 c) {
  return __builtin_amdgcn_wmma_f32_16x16x32_bf16(false, a, false, b,
                                                 (short)0, c, false, false);
}

__device__ __forceinline__ float lrelu(float x) { return x >= 0.f ? x : 0.01f * x; }

// ---------------------------------------------------------------- weights cvt
__global__ __launch_bounds__(256) void cvt_bf16_kernel(const float* __restrict__ in,
                                                       __bf16* __restrict__ out, int n) {
  int i = blockIdx.x * 256 + threadIdx.x;
  if (i < n) out[i] = (__bf16)in[i];
}

// -------------------------------------------------------- TDM weight staging
// Load a 256(M) x 32(K) bf16 tile of a row-major [256][256] weight matrix into
// LDS laid out [256][40] (TDM pad: every 16 DWORDs insert 4 DWORDs -> 80B rows).
__device__ __forceinline__ void stage_w_tile(const __bf16* __restrict__ Wsrc, int kb,
                                             __bf16 (*wT)[40], int tid, int wave) {
#if __has_builtin(__builtin_amdgcn_tensor_load_to_lds)
  if (wave == 0) {
    unsigned long long ga = (unsigned long long)(Wsrc + kb);
    unsigned ldsOff = (unsigned)(unsigned long long)&wT[0][0];
    u32x4 g0;
    g0[0] = 1u;                                        // count=1, user mode
    g0[1] = ldsOff;                                    // lds_addr
    g0[2] = (unsigned)(ga & 0xffffffffu);              // global_addr[31:0]
    g0[3] = (unsigned)((ga >> 32) & 0x1ffffffu)        // global_addr[56:32]
            | (2u << 30);                              // type=2 (image)
    i32x8 g1;
    g1[0] = (int)((1u << 16)                           // data_size = 2B
                  | (1u << 20)                         // pad_enable
                  | (3u << 22)                         // pad_interval: 16 DWORDs
                  | (3u << 25));                       // pad_amount: 4 DWORDs
    g1[1] = (int)(256u << 16);                         // tensor_dim0 = 256 (lo16)
    g1[2] = (int)(256u << 16);                         // dim0 hi16=0 | tensor_dim1=256
    g1[3] = (int)(32u << 16);                          // dim1 hi16=0 | tile_dim0=32
    g1[4] = (int)256u;                                 // tile_dim1=256 | tile_dim2=0
    g1[5] = (int)256u;                                 // tensor_dim0_stride = 256
    g1[6] = 0;                                         // stride0 hi | stride1 lo
    g1[7] = 0;
    i32x4 gz = {0, 0, 0, 0};
#if __clang_major__ >= 23
    i32x8 gz8 = {0, 0, 0, 0, 0, 0, 0, 0};
    __builtin_amdgcn_tensor_load_to_lds(g0, g1, gz, gz, gz8, 0);
#else
    __builtin_amdgcn_tensor_load_to_lds(g0, g1, gz, gz, 0);
#endif
#if __has_builtin(__builtin_amdgcn_s_wait_tensorcnt)
    __builtin_amdgcn_s_wait_tensorcnt(0);
#endif
  }
#else
  (void)wave;
  for (int i = tid; i < 1024; i += 256) {              // 1024 chunks of 8 bf16
    int row = i >> 2, kq = (i & 3) * 8;
    *(u32x4*)&wT[row][kq] = *(const u32x4*)(Wsrc + (long long)row * CH + kb + kq);
  }
#endif
}

// ------------------------------------------------- generic 256xK x KxN GEMM
// Out[b][m][n] = sum_k A[b][m][k] * X[b][k][n] (+ bias[m]); M=K=256, N=9216.
// Block 256 thr (8 waves), tile 256(M) x 64(N); double-buffered B staging.
// Fragment ds_loads are issued BEFORE the next-buffer ds_stores so the WMMA
// dscnt wait need not drain the staging stores (DS retires in order).
template <typename TX, bool HAS_BIAS>
__global__ __launch_bounds__(256) void gemm256_kernel(
    const TX* __restrict__ X, long long xStride,
    const __bf16* __restrict__ A, long long aStride,
    const float* __restrict__ bias,
    __bf16* __restrict__ Out)
{
  __shared__ alignas(16) __bf16 bT[2][64][40];   // [n][k] transposed, padded rows
  const int b   = blockIdx.y;
  const int n0g = blockIdx.x * 64;
  const TX*     Xb = X + (long long)b * xStride;
  const __bf16* Ab = A + (long long)b * aStride;
  __bf16*       Ob = Out + (long long)b * ((long long)CH * NSP);
  const int tid = threadIdx.x, lane = tid & 31, wave = tid >> 5;
  const int m0 = wave * 32;

  auto stage = [&](int buf, int kb) {
#pragma unroll
    for (int s = 0; s < 8; ++s) {
      int idx = tid + s * 256;
      int k = idx >> 6, n = idx & 63;                 // consecutive tid -> n (coalesced)
      bT[buf][n][k] = (__bf16)(float)Xb[(long long)(kb + k) * NSP + n0g + n];
    }
  };

  f32x8 acc[2][4] = {};
  stage(0, 0);
  __syncthreads();
  for (int kb = 0; kb < CH; kb += 32) {
    const int cur = (kb >> 5) & 1;
    // fragment loads first (global A, LDS B)
    bf16x16 a0 = load_frag_a(Ab, CH, m0, kb, lane);
    bf16x16 a1 = load_frag_a(Ab, CH, m0 + 16, kb, lane);
    bf16x16 bfr[4];
#pragma unroll
    for (int t = 0; t < 4; ++t)
      bfr[t] = load_frag_b(&bT[cur][0][0], 40, t * 16, 0, lane);
    // then stage next K-slab (overlaps the WMMAs below)
    if (kb + 32 < CH) stage(cur ^ 1, kb + 32);
#pragma unroll
    for (int t = 0; t < 4; ++t) {
      acc[0][t] = wmma_bf16(a0, bfr[t], acc[0][t]);
      acc[1][t] = wmma_bf16(a1, bfr[t], acc[1][t]);
    }
    __syncthreads();
  }
  const int hf = lane >> 4, cL = lane & 15;
#pragma unroll
  for (int mi = 0; mi < 2; ++mi)
#pragma unroll
    for (int t = 0; t < 4; ++t)
#pragma unroll
      for (int i = 0; i < 8; ++i) {
        int row = m0 + mi * 16 + hf * 8 + i;
        float v = acc[mi][t][i];
        if (HAS_BIAS) v += bias[row];
        Ob[(long long)row * NSP + n0g + t * 16 + cL] = (__bf16)v;
      }
}

// --------------------------------------------------------------- Q.K^T scores
// S[b][c][d] = (1/96) * sum_n Q[b][c][n] * K[b][d][n].
// Block tile 64x64; Q/K K-slabs staged in LDS, shared by all 8 waves,
// double-buffered; fragment ds_loads issued before staging ds_stores.
__global__ __launch_bounds__(256) void scores_kernel(
    const __bf16* __restrict__ Q, const __bf16* __restrict__ Kt,
    float* __restrict__ S)
{
  __shared__ alignas(16) __bf16 qT[2][64][40];   // [m][k] rows (A-style)
  __shared__ alignas(16) __bf16 kT[2][64][40];   // [n][k] rows (B-style)
  const int b    = blockIdx.y;
  const int mBlk = (blockIdx.x & 3) * 64;
  const int nBlk = (blockIdx.x >> 2) * 64;
  const int tid = threadIdx.x, lane = tid & 31, wave = tid >> 5;
  const __bf16* Qb = Q  + (long long)b * CH * NSP + (long long)mBlk * NSP;
  const __bf16* Kb = Kt + (long long)b * CH * NSP + (long long)nBlk * NSP;
  const int srow = tid >> 2, skq = (tid & 3) * 8;  // 4 threads/row, 8 bf16 each

  auto stage = [&](int buf, int kb) {
    *(u32x4*)&qT[buf][srow][skq] = *(const u32x4*)(Qb + (long long)srow * NSP + kb + skq);
    *(u32x4*)&kT[buf][srow][skq] = *(const u32x4*)(Kb + (long long)srow * NSP + kb + skq);
  };

  const int m0 = (wave & 3) * 16;
  const int n0 = (wave >> 2) * 32;
  f32x8 acc0 = {}, acc1 = {};
  stage(0, 0);
  __syncthreads();
  for (int kb = 0; kb < NSP; kb += 32) {
    const int cur = (kb >> 5) & 1;
    // fragment ds_loads from current buffer first
    bf16x16 a  = load_frag_a(&qT[cur][0][0], 40, m0, 0, lane);
    bf16x16 b0 = load_frag_b(&kT[cur][0][0], 40, n0, 0, lane);
    bf16x16 b1 = load_frag_b(&kT[cur][0][0], 40, n0 + 16, 0, lane);
    // then stage the next slab (global b128 + ds_store overlap the WMMAs)
    if (kb + 32 < NSP) {
      __builtin_prefetch(Qb + (long long)srow * NSP + kb + 96, 0, 3);
      __builtin_prefetch(Kb + (long long)srow * NSP + kb + 96, 0, 3);
      stage(cur ^ 1, kb + 32);
    }
    acc0 = wmma_bf16(a, b0, acc0);
    acc1 = wmma_bf16(a, b1, acc1);
    __syncthreads();
  }
  const int hf = lane >> 4, cL = lane & 15;
  float* Sb = S + (long long)b * CH * CH;
  const float sc = 1.0f / 96.0f;   // sqrt(9216) == 96
#pragma unroll
  for (int i = 0; i < 8; ++i) {
    int row = mBlk + m0 + hf * 8 + i;
    Sb[row * CH + nBlk + n0 + cL]      = acc0[i] * sc;
    Sb[row * CH + nBlk + n0 + 16 + cL] = acc1[i] * sc;
  }
}

// ------------------------------------------------------------------- softmax
__global__ __launch_bounds__(256) void softmax_kernel(const float* __restrict__ S,
                                                      __bf16* __restrict__ attn)
{
  const int wave = threadIdx.x >> 5, lane = threadIdx.x & 31;
  const long long row = (long long)blockIdx.x * 8 + wave;
  const float* s = S + row * CH;
  float v[8];
  float mx = -1e30f;
#pragma unroll
  for (int i = 0; i < 8; ++i) { v[i] = s[lane + i * 32]; mx = fmaxf(mx, v[i]); }
  for (int o = 16; o > 0; o >>= 1) mx = fmaxf(mx, __shfl_xor(mx, o, 32));
  float sum = 0.f;
#pragma unroll
  for (int i = 0; i < 8; ++i) { v[i] = __expf(v[i] - mx); sum += v[i]; }
  for (int o = 16; o > 0; o >>= 1) sum += __shfl_xor(sum, o, 32);
  const float r = 1.0f / sum;
  __bf16* a = attn + row * CH;
#pragma unroll
  for (int i = 0; i < 8; ++i) a[lane + i * 32] = (__bf16)(v[i] * r);
}

// -------------------------------- fused shuffle + BN + lrelu + conv1 + conv2
// pre[b,c,p] = ao[b, p%256, c*36 + p/256]  (permute(0,2,1).view; 9216%256==0).
// Whole chain is column-local -> one block per 32-column spatial tile.
// Weight K-tiles staged into LDS via the Tensor Data Mover (padded rows).
__global__ __launch_bounds__(256) void out_fused_kernel(
    const __bf16* __restrict__ ao,
    const float* __restrict__ g,  const float* __restrict__ be,
    const float* __restrict__ mu, const float* __restrict__ var,
    const __bf16* __restrict__ W1, const float* __restrict__ b1,
    const __bf16* __restrict__ W2, const float* __restrict__ b2,
    float* __restrict__ out)
{
  __shared__ alignas(16) __bf16 xT[32][264];    // [col][k], 16B-multiple rows
  __shared__ alignas(16) __bf16 hT[32][264];
  __shared__ alignas(16) __bf16 wT[256][40];    // [m][k] weight tile (TDM-padded)
  const int b  = blockIdx.y;
  const int p0 = blockIdx.x * 32;
  const int tid = threadIdx.x, lane = tid & 31, wave = tid >> 5;
  const __bf16* aob = ao + (long long)b * CH * NSP;

  // gather-shuffle + BN + leaky relu into LDS
  for (int idx = tid; idx < 32 * CH; idx += 256) {
    int j = idx & 31, c = idx >> 5;
    int p = p0 + j;
    float x  = (float)aob[(long long)(p & 255) * NSP + c * 36 + (p >> 8)];
    float sc = g[c] * rsqrtf(var[c] + 1e-4f);
    xT[j][c] = (__bf16)lrelu((x - mu[c]) * sc + be[c]);
  }
  __syncthreads();

  const int m0 = (wave & 3) * 64;       // wave tile: 64(M) x 16(N)
  const int n0 = (wave >> 2) * 16;
  const int hf = lane >> 4, cL = lane & 15;

  // conv1: h = lrelu(W1 * x + b1)
  f32x8 acc[4] = {};
  for (int kb = 0; kb < CH; kb += 32) {
    stage_w_tile(W1, kb, wT, tid, wave);
    __syncthreads();
    bf16x16 bf = load_frag_b(&xT[0][0], 264, n0, kb, lane);
#pragma unroll
    for (int t = 0; t < 4; ++t) {
      bf16x16 a = load_frag_a(&wT[0][0], 40, m0 + t * 16, 0, lane);
      acc[t] = wmma_bf16(a, bf, acc[t]);
    }
    __syncthreads();
  }
#pragma unroll
  for (int t = 0; t < 4; ++t)
#pragma unroll
    for (int i = 0; i < 8; ++i) {
      int row = m0 + t * 16 + hf * 8 + i;
      hT[n0 + cL][row] = (__bf16)lrelu(acc[t][i] + b1[row]);
    }
  __syncthreads();

  // conv2: out = W2 * h + b2
  f32x8 acc2[4] = {};
  for (int kb = 0; kb < CH; kb += 32) {
    stage_w_tile(W2, kb, wT, tid, wave);
    __syncthreads();
    bf16x16 bf = load_frag_b(&hT[0][0], 264, n0, kb, lane);
#pragma unroll
    for (int t = 0; t < 4; ++t) {
      bf16x16 a = load_frag_a(&wT[0][0], 40, m0 + t * 16, 0, lane);
      acc2[t] = wmma_bf16(a, bf, acc2[t]);
    }
    __syncthreads();
  }
  float* ob = out + (long long)b * CH * NSP;
#pragma unroll
  for (int t = 0; t < 4; ++t)
#pragma unroll
    for (int i = 0; i < 8; ++i) {
      int row = m0 + t * 16 + hf * 8 + i;
      ob[(long long)row * NSP + p0 + n0 + cL] = acc2[t][i] + b2[row];
    }
}

// ------------------------------------------------------------------ launcher
extern "C" void kernel_launch(void* const* d_in, const int* in_sizes, int n_in,
                              void* d_out, int out_size, void* d_ws, size_t ws_size,
                              hipStream_t stream)
{
  (void)in_sizes; (void)n_in; (void)out_size; (void)ws_size;
  const float* q   = (const float*)d_in[0];
  const float* v   = (const float*)d_in[1];
  const float* Wq  = (const float*)d_in[2];
  const float* bq  = (const float*)d_in[3];
  const float* Wk  = (const float*)d_in[4];
  const float* bk  = (const float*)d_in[5];
  const float* Wv  = (const float*)d_in[6];
  const float* bv  = (const float*)d_in[7];
  const float* bng = (const float*)d_in[8];
  const float* bnb = (const float*)d_in[9];
  const float* bnm = (const float*)d_in[10];
  const float* bnv = (const float*)d_in[11];
  const float* Wo1 = (const float*)d_in[12];
  const float* bo1 = (const float*)d_in[13];
  const float* Wo2 = (const float*)d_in[14];
  const float* bo2 = (const float*)d_in[15];
  float* out = (float*)d_out;

  char* ws = (char*)d_ws;
  size_t off = 0;
  auto take = [&](size_t bytes) -> void* {
    void* p = ws + off;
    off += (bytes + 255) & ~(size_t)255;
    return p;
  };
  const long long WB = (long long)CH * CH;    // 65536
  const long long XB = (long long)CH * NSP;   // per-batch activation elems

  __bf16* wq_bf    = (__bf16*)take(WB * 2);
  __bf16* wk_bf    = (__bf16*)take(WB * 2);
  __bf16* wv_bf    = (__bf16*)take(WB * 2);
  __bf16* w1_bf    = (__bf16*)take(WB * 2);
  __bf16* w2_bf    = (__bf16*)take(WB * 2);
  __bf16* query_bf = (__bf16*)take((long long)NB * XB * 2);
  __bf16* key_bf   = (__bf16*)take((long long)NB * XB * 2);
  __bf16* value_bf = (__bf16*)take((long long)NB * XB * 2);
  __bf16* ao_bf    = (__bf16*)take((long long)NB * XB * 2);
  float*  scores   = (float*)take((long long)NB * WB * 4);
  __bf16* attn_bf  = (__bf16*)take((long long)NB * WB * 2);

  dim3 blk(256);
  cvt_bf16_kernel<<<dim3(256), blk, 0, stream>>>(Wq,  wq_bf, (int)WB);
  cvt_bf16_kernel<<<dim3(256), blk, 0, stream>>>(Wk,  wk_bf, (int)WB);
  cvt_bf16_kernel<<<dim3(256), blk, 0, stream>>>(Wv,  wv_bf, (int)WB);
  cvt_bf16_kernel<<<dim3(256), blk, 0, stream>>>(Wo1, w1_bf, (int)WB);
  cvt_bf16_kernel<<<dim3(256), blk, 0, stream>>>(Wo2, w2_bf, (int)WB);

  // Q = Wq*q + bq ; K = Wk*v + bk ; V = Wv*q + bv
  gemm256_kernel<float, true><<<dim3(NSP / 64, NB), blk, 0, stream>>>(q, XB, wq_bf, 0, bq, query_bf);
  gemm256_kernel<float, true><<<dim3(NSP / 64, NB), blk, 0, stream>>>(v, XB, wk_bf, 0, bk, key_bf);
  gemm256_kernel<float, true><<<dim3(NSP / 64, NB), blk, 0, stream>>>(q, XB, wv_bf, 0, bv, value_bf);

  scores_kernel<<<dim3(16, NB), blk, 0, stream>>>(query_bf, key_bf, scores);
  softmax_kernel<<<dim3(NB * CH / 8), blk, 0, stream>>>(scores, attn_bf);

  // ao = attn * value  (A per-batch: stride WB)
  gemm256_kernel<__bf16, false><<<dim3(NSP / 64, NB), blk, 0, stream>>>(value_bf, XB, attn_bf, WB, nullptr, ao_bf);

  out_fused_kernel<<<dim3(NSP / 32, NB), blk, 0, stream>>>(ao_bf, bng, bnb, bnm, bnv,
                                                           w1_bf, bo1, w2_bf, bo2, out);
}
